// NonLocal_46943992545435
// MI455X (gfx1250) — matrix operand order repaired and checked by reading, not compiled, not run on hardware
//
#include <hip/hip_runtime.h>
#include <hip/hip_fp16.h>
#include <cstdint>

typedef __attribute__((ext_vector_type(16))) _Float16 v16h;
typedef __attribute__((ext_vector_type(8)))  _Float16 v8h;
typedef __attribute__((ext_vector_type(8)))  float    v8f;

#define BD   4
#define CD   256
#define C2D  128
#define ND   4096
#define LOG2E 1.4426950408889634f

static __device__ __forceinline__ v8h load8h(const _Float16* p) {
  return *(const v8h*)p;
}
static __device__ __forceinline__ v16h cat8(v8h lo, v8h hi) {
  v16h r;
#pragma unroll
  for (int i = 0; i < 8; i++) { r[i] = lo[i]; r[i + 8] = hi[i]; }
  return r;
}
static __device__ __forceinline__ v16h load16h(const _Float16* p) {
  return cat8(*(const v8h*)p, *(const v8h*)(p + 8));
}
static __device__ __forceinline__ v8f wmma16(v16h a, v16h b, v8f c) {
  // D = A(16x32 f16) * B(32x16 f16) + C(16x16 f32)
  return __builtin_amdgcn_wmma_f32_16x16x32_f16(false, a, false, b, (short)0, c,
                                                false, false);
}

// ---------------------------------------------------------------------------
// 1) x[b][c][n] f32  ->  xh[b][n][c] f16  (LDS 32x32 tile transpose)
// ---------------------------------------------------------------------------
__global__ void pack_x(const float* __restrict__ x, _Float16* __restrict__ xh) {
  __shared__ float tile[32][33];
  const int b  = blockIdx.z;
  const int c0 = blockIdx.y * 32;
  const int n0 = blockIdx.x * 32;
  const int tx = threadIdx.x & 31;
  const int ty = threadIdx.x >> 5;   // 0..7
#pragma unroll
  for (int i = 0; i < 4; i++) {
    int c = c0 + ty + i * 8;
    tile[ty + i * 8][tx] = x[((size_t)b * CD + c) * ND + n0 + tx];
  }
  __syncthreads();
#pragma unroll
  for (int i = 0; i < 4; i++) {
    int n = n0 + ty + i * 8;
    xh[((size_t)b * ND + n) * CD + c0 + tx] = (_Float16)tile[tx][ty + i * 8];
  }
}

// ---------------------------------------------------------------------------
// 2) generic f32 -> f16 convert (weights)
// ---------------------------------------------------------------------------
__global__ void cvt_f16(const float* __restrict__ s, _Float16* __restrict__ d, int n) {
  int i = blockIdx.x * blockDim.x + threadIdx.x;
  if (i < n) d[i] = (_Float16)s[i];
}

// ---------------------------------------------------------------------------
// 3) theta/phi projections.  D layout: M = out-channel o, N = spatial n.
//    A = W rows (o x c), B = xh rows (c x n via [n][c] storage).
//    Output stored [b][n][c2] (row-major c2) -> per-lane 16B store.
// ---------------------------------------------------------------------------
__global__ void proj_theta_phi(const _Float16* __restrict__ xh,
                               const _Float16* __restrict__ whT,
                               const _Float16* __restrict__ whP,
                               const float* __restrict__ bT,
                               const float* __restrict__ bP,
                               _Float16* __restrict__ thetaH,
                               _Float16* __restrict__ phiH) {
  const int lane = threadIdx.x & 31;
  const int wid  = blockIdx.x * (blockDim.x >> 5) + (threadIdx.x >> 5);
  const int nt = wid & 255;
  const int ot = (wid >> 8) & 7;
  const int which = (wid >> 11) & 1;
  const int b = wid >> 12;
  const _Float16* wh  = which ? whP : whT;
  const float* bias   = which ? bP : bT;
  _Float16* dst       = which ? phiH : thetaH;
  const int n0 = nt * 16, o0 = ot * 16;
  const int ll = lane & 15, lh = lane >> 4;

  v8f acc = {};
  const _Float16* wrow = wh + (size_t)(o0 + ll) * CD;
  const _Float16* xrow = xh + ((size_t)b * ND + n0 + ll) * CD;
#pragma unroll
  for (int kc = 0; kc < 8; kc++) {
    v16h a  = load16h(wrow + kc * 32 + lh * 16);
    v16h bf = load16h(xrow + kc * 32 + lh * 16);
    acc = wmma16(a, bf, acc);
  }
  // lane holds o = o0 + lh*8 + r, n = n0 + ll
  v8h oh;
#pragma unroll
  for (int r = 0; r < 8; r++)
    oh[r] = (_Float16)(acc[r] + bias[o0 + lh * 8 + r]);
  *(v8h*)(dst + ((size_t)b * ND + n0 + ll) * C2D + o0 + lh * 8) = oh;
}

// ---------------------------------------------------------------------------
// 4) g projection.  D layout: M = n, N = o.  Output gT[b][c2][n].
// ---------------------------------------------------------------------------
__global__ void proj_g(const _Float16* __restrict__ xh,
                       const _Float16* __restrict__ whG,
                       const float* __restrict__ bG,
                       _Float16* __restrict__ gT) {
  const int lane = threadIdx.x & 31;
  const int wid  = blockIdx.x * (blockDim.x >> 5) + (threadIdx.x >> 5);
  const int nt = wid & 255;
  const int ot = (wid >> 8) & 7;
  const int b  = wid >> 11;
  const int n0 = nt * 16, o0 = ot * 16;
  const int ll = lane & 15, lh = lane >> 4;

  v8f acc = {};
  const _Float16* xrow = xh + ((size_t)b * ND + n0 + ll) * CD;
  const _Float16* wrow = whG + (size_t)(o0 + ll) * CD;
#pragma unroll
  for (int kc = 0; kc < 8; kc++) {
    v16h a  = load16h(xrow + kc * 32 + lh * 16);
    v16h bf = load16h(wrow + kc * 32 + lh * 16);
    acc = wmma16(a, bf, acc);
  }
  // lane holds n = n0 + lh*8 + r, o = o0 + ll
  const float bv = bG[o0 + ll];
  v8h oh;
#pragma unroll
  for (int r = 0; r < 8; r++) oh[r] = (_Float16)(acc[r] + bv);
  *(v8h*)(gT + ((size_t)b * C2D + o0 + ll) * ND + n0 + lh * 8) = oh;
}

// ---------------------------------------------------------------------------
// 5) Flash attention.  One wave = 16 queries x all 4096 keys, C2=128 output.
//    S gemm:  A = theta(16x32 chunks), B = phi^T slice  -> S (M=q, N=key)
//    O gemm:  A = gT(16ch x 32key),    B = P^T from LDS -> O^T (M=ch, N=q)
// ---------------------------------------------------------------------------
#define AW 8   // waves per block
__global__ void attn(const _Float16* __restrict__ thetaH,
                     const _Float16* __restrict__ phiH,
                     const _Float16* __restrict__ gT,
                     _Float16* __restrict__ y) {
  __shared__ _Float16 sP[AW][16 * 32];
  __shared__ float    sS[AW][16];
  const int lane = threadIdx.x & 31;
  const int w    = threadIdx.x >> 5;
  const int wid  = blockIdx.x * AW + w;       // 0..1023
  const int b  = wid >> 8;
  const int q0 = (wid & 255) * 16;
  const int ll = lane & 15, lh = lane >> 4;

  // theta A fragments for the whole tile (16 q x 128 c)
  const _Float16* trow = thetaH + ((size_t)b * ND + q0 + ll) * C2D;
  v16h thA[4];
#pragma unroll
  for (int kc = 0; kc < 4; kc++)
    thA[kc] = load16h(trow + kc * 32 + lh * 16);

  v8f acc[8];
#pragma unroll
  for (int ot = 0; ot < 8; ot++)
#pragma unroll
    for (int r = 0; r < 8; r++) acc[ot][r] = 0.0f;

  float mrow[8], lrow[8];
#pragma unroll
  for (int r = 0; r < 8; r++) { mrow[r] = -3.0e38f; lrow[r] = 0.0f; }

  const _Float16* phiB = phiH + (size_t)b * ND * C2D;
  const _Float16* gB   = gT + (size_t)b * C2D * ND;

  for (int kb = 0; kb < ND; kb += 32) {
    // ---- S = theta * phi^T for 32 keys (two 16-key column tiles) ----
    v8f S0 = {}, S1 = {};
    const _Float16* p0r = phiB + (size_t)(kb + ll) * C2D + lh * 16;
    const _Float16* p1r = phiB + (size_t)(kb + 16 + ll) * C2D + lh * 16;
#pragma unroll
    for (int kc = 0; kc < 4; kc++) {
      S0 = wmma16(thA[kc], load16h(p0r + kc * 32), S0);
      S1 = wmma16(thA[kc], load16h(p1r + kc * 32), S1);
    }
    if (kb + 32 < ND)
      __builtin_prefetch(phiB + (size_t)(kb + 32 + ll) * C2D, 0, 1);

    // ---- online softmax in S layout: lane holds q = lh*8+r, key = ll(+16)
    float p0[8], p1[8], sc[8];
#pragma unroll
    for (int r = 0; r < 8; r++) {
      float mx = fmaxf(S0[r], S1[r]);
#pragma unroll
      for (int m = 1; m < 16; m <<= 1) mx = fmaxf(mx, __shfl_xor(mx, m, 32));
      float mnew = fmaxf(mrow[r], mx);
      sc[r] = exp2f((mrow[r] - mnew) * LOG2E);
      p0[r] = exp2f((S0[r] - mnew) * LOG2E);
      p1[r] = exp2f((S1[r] - mnew) * LOG2E);
      float rs = p0[r] + p1[r];
#pragma unroll
      for (int m = 1; m < 16; m <<= 1) rs += __shfl_xor(rs, m, 32);
      lrow[r] = lrow[r] * sc[r] + rs;
      mrow[r] = mnew;
    }

    // broadcast per-query rescale factors through LDS (same-wave, DS in-order)
    if (ll == 0) {
#pragma unroll
      for (int r = 0; r < 8; r++) sS[w][lh * 8 + r] = sc[r];
    }
    // P tile (16q x 32key) to LDS row-major for B-fragment reload
#pragma unroll
    for (int r = 0; r < 8; r++) {
      int q = lh * 8 + r;
      sP[w][q * 32 + ll]      = (_Float16)p0[r];
      sP[w][q * 32 + 16 + ll] = (_Float16)p1[r];
    }
    __builtin_amdgcn_wave_barrier();

    const float s = sS[w][ll];        // lane's query in O^T layout = ll
#pragma unroll
    for (int ot = 0; ot < 8; ot++)
#pragma unroll
      for (int r = 0; r < 8; r++) acc[ot][r] *= s;

    // ---- O^T += gT * P^T ----
    v16h Pb = load16h(&sP[w][ll * 32 + lh * 16]);
#pragma unroll
    for (int ot = 0; ot < 8; ot++) {
      const _Float16* gr = gB + (size_t)(ot * 16 + ll) * ND + kb + lh * 16;
      v16h ga = load16h(gr);
      acc[ot] = wmma16(ga, Pb, acc[ot]);
    }
  }

  // ---- normalize and store y[b][n][c2] ----
  if (ll == 0) {
#pragma unroll
    for (int r = 0; r < 8; r++) sS[w][lh * 8 + r] = 1.0f / lrow[r];
  }
  __builtin_amdgcn_wave_barrier();
  const float inv = sS[w][ll];
  _Float16* yp = y + ((size_t)b * ND + q0 + ll) * C2D + lh * 8;
#pragma unroll
  for (int ot = 0; ot < 8; ot++) {
    v8h oh;
#pragma unroll
    for (int r = 0; r < 8; r++) oh[r] = (_Float16)(acc[ot][r] * inv);
    *(v8h*)(yp + ot * 16) = oh;
  }
}

// ---------------------------------------------------------------------------
// 6) out = x + W_final * y + b_final.   D: M = n, N = o.
// ---------------------------------------------------------------------------
__global__ void final_proj(const _Float16* __restrict__ y,
                           const _Float16* __restrict__ whF,
                           const float* __restrict__ bF,
                           const float* __restrict__ x,
                           float* __restrict__ out) {
  const int lane = threadIdx.x & 31;
  const int wid  = blockIdx.x * (blockDim.x >> 5) + (threadIdx.x >> 5);
  const int nt = wid & 255;
  const int ot = (wid >> 8) & 15;
  const int b  = wid >> 12;
  const int n0 = nt * 16, o0 = ot * 16;
  const int ll = lane & 15, lh = lane >> 4;

  v8f acc = {};
  const _Float16* yrow = y + ((size_t)b * ND + n0 + ll) * C2D;
  const _Float16* wrow = whF + (size_t)(o0 + ll) * C2D;
#pragma unroll
  for (int kc = 0; kc < 4; kc++) {
    v16h a  = load16h(yrow + kc * 32 + lh * 16);
    v16h bf = load16h(wrow + kc * 32 + lh * 16);
    acc = wmma16(a, bf, acc);
  }
  // lane holds n = n0 + lh*8 + r, o = o0 + ll
  const float bias = bF[o0 + ll];
  const float* xr = x + ((size_t)b * CD + o0 + ll) * ND + n0 + lh * 8;
  float* op = out + ((size_t)b * CD + o0 + ll) * ND + n0 + lh * 8;
#pragma unroll
  for (int r = 0; r < 8; r++) op[r] = xr[r] + acc[r] + bias;
}

// ---------------------------------------------------------------------------
extern "C" void kernel_launch(void* const* d_in, const int* in_sizes, int n_in,
                              void* d_out, int out_size, void* d_ws, size_t ws_size,
                              hipStream_t stream) {
  const float* x       = (const float*)d_in[0];
  const float* w_theta = (const float*)d_in[1];
  const float* b_theta = (const float*)d_in[2];
  const float* w_phi   = (const float*)d_in[3];
  const float* b_phi   = (const float*)d_in[4];
  const float* w_g     = (const float*)d_in[5];
  const float* b_g     = (const float*)d_in[6];
  const float* w_final = (const float*)d_in[7];
  const float* b_final = (const float*)d_in[8];
  float* out = (float*)d_out;

  char* ws = (char*)d_ws;
  size_t off = 0;
  _Float16* xh  = (_Float16*)(ws + off); off += (size_t)BD * ND * CD * 2;   // 8 MB
  _Float16* whT = (_Float16*)(ws + off); off += (size_t)C2D * CD * 2;
  _Float16* whP = (_Float16*)(ws + off); off += (size_t)C2D * CD * 2;
  _Float16* whG = (_Float16*)(ws + off); off += (size_t)C2D * CD * 2;
  _Float16* whF = (_Float16*)(ws + off); off += (size_t)CD * C2D * 2;
  _Float16* thetaH = (_Float16*)(ws + off); off += (size_t)BD * ND * C2D * 2;
  _Float16* phiH   = (_Float16*)(ws + off); off += (size_t)BD * ND * C2D * 2;
  _Float16* gTm    = (_Float16*)(ws + off); off += (size_t)BD * C2D * ND * 2;
  _Float16* yv     = (_Float16*)(ws + off); off += (size_t)BD * ND * C2D * 2;
  (void)ws_size; (void)in_sizes; (void)n_in; (void)out_size;

  pack_x<<<dim3(ND / 32, CD / 32, BD), 256, 0, stream>>>(x, xh);

  const int wn = C2D * CD;  // 32768 elements per weight matrix
  cvt_f16<<<(wn + 255) / 256, 256, 0, stream>>>(w_theta, whT, wn);
  cvt_f16<<<(wn + 255) / 256, 256, 0, stream>>>(w_phi,   whP, wn);
  cvt_f16<<<(wn + 255) / 256, 256, 0, stream>>>(w_g,     whG, wn);
  cvt_f16<<<(wn + 255) / 256, 256, 0, stream>>>(w_final, whF, wn);

  // 4(b) * 2(theta/phi) * 8(o-tiles) * 256(n-tiles) = 16384 waves, 8 per block
  proj_theta_phi<<<2048, 256, 0, stream>>>(xh, whT, whP, b_theta, b_phi,
                                           thetaH, phiH);
  // 4 * 8 * 256 = 8192 waves
  proj_g<<<1024, 256, 0, stream>>>(xh, whG, b_g, gTm);

  // 4 * 256 query tiles = 1024 waves, AW=8 per block
  attn<<<1024 / AW, 32 * AW, 0, stream>>>(thetaH, phiH, gTm, yv);

  // 4 * 16 * 256 = 16384 waves
  final_proj<<<2048, 256, 0, stream>>>(yv, whF, b_final, x, out);
}
